// CrossModalAttention_45062796870248
// MI455X (gfx1250) — matrix operand order repaired and checked by reading, not compiled, not run on hardware
//
#include <hip/hip_runtime.h>
#include <hip/hip_fp16.h>

typedef _Float16 half8_t  __attribute__((ext_vector_type(8)));
typedef _Float16 half16_t __attribute__((ext_vector_type(16)));
typedef float    float8_t __attribute__((ext_vector_type(8)));

#define NB 32
#define NC 256
#define NH 64
#define NW 64
#define NT 64          // tokens = 8*8
#define LDP 264        // padded row length (halves) for [64][256] tiles
#define LDV 72         // padded row length (halves) for [256][64] / [64][64] tiles
#define LDAF 68        // padded row length (floats) for scores

// ---------------- WMMA fragment helpers (wave32, 16x16x32 f16) ----------------
// A fragment from row-major [16][ld] f16 tile (M x K):
//   lane<16: row=lane, halves = K[0:8] ++ K[16:24]; lane>=16: row=lane-16, K[8:16] ++ K[24:32]
__device__ __forceinline__ half16_t frag_a(const _Float16* p, int ld) {
  int lane = threadIdx.x & 31;
  const _Float16* r = p + (lane & 15) * ld + ((lane >> 4) << 3);
  half8_t lo = *reinterpret_cast<const half8_t*>(r);
  half8_t hi = *reinterpret_cast<const half8_t*>(r + 16);
  return __builtin_shufflevector(lo, hi, 0,1,2,3,4,5,6,7,8,9,10,11,12,13,14,15);
}
// B fragment from [N=16][ld] f16 storage (i.e. B^T row-major):
//   lane<16: col=lane, halves = K[0:8] ++ K[16:24]; lane>=16: col=lane-16, K[8:16] ++ K[24:32]
__device__ __forceinline__ half16_t frag_b(const _Float16* p, int ld) {
  int lane = threadIdx.x & 31;
  const _Float16* r = p + (lane & 15) * ld + ((lane >> 4) << 3);
  half8_t lo = *reinterpret_cast<const half8_t*>(r);
  half8_t hi = *reinterpret_cast<const half8_t*>(r + 16);
  return __builtin_shufflevector(lo, hi, 0,1,2,3,4,5,6,7,8,9,10,11,12,13,14,15);
}
__device__ __forceinline__ float8_t wmma_f16(half16_t a, half16_t b, float8_t c) {
  return __builtin_amdgcn_wmma_f32_16x16x32_f16(false, a, false, b, (short)0, c, false, false);
}

// ---------------- K0: f32 -> f16 weight conversion ----------------
__global__ void __launch_bounds__(256)
cvt_f32_to_f16(const float* __restrict__ src, _Float16* __restrict__ dst, int n) {
  int i = blockIdx.x * 256 + threadIdx.x;
  if (i < n) dst[i] = (_Float16)src[i];
}

// ---------------- K1: 8x8 adaptive avg pool, output transposed [b][t][c] f16 ----
__global__ void __launch_bounds__(64)
pool_kernel(const float* __restrict__ F_rgb, const float* __restrict__ F_d,
            _Float16* __restrict__ Rt, _Float16* __restrict__ Dt) {
  const int bc = blockIdx.x;                         // b*NC + c
  const float* src = (blockIdx.y == 0 ? F_rgb : F_d) + (size_t)bc * (NH * NW);
  _Float16* dst = (blockIdx.y == 0 ? Rt : Dt);
  const int t = threadIdx.x;                         // token 0..63
  const int ty = t >> 3, tx = t & 7;
  float s = 0.f;
#pragma unroll
  for (int ry = 0; ry < 8; ++ry) {
    const float* row = src + (ty * 8 + ry) * NW + tx * 8;
#pragma unroll
    for (int rx = 0; rx < 8; ++rx) s += row[rx];
  }
  const int b = bc >> 8, c = bc & 255;
  dst[(size_t)b * (NT * NC) + t * NC + c] = (_Float16)(s * (1.f / 64.f));
}

// ---------------- K2: fused attention core, one block per batch --------------
// LDS (halves): Qt[64][264] | Kt[64][264] | {Rt[64][264] -> Vm[256][72]} |
//               {Dt[64][264] -> Af(f32)[64][68] + As[64][72]}
__global__ void __launch_bounds__(256)
attn_kernel(const _Float16* __restrict__ Rt_g, const _Float16* __restrict__ Dt_g,
            const _Float16* __restrict__ Wq, const _Float16* __restrict__ Wk,
            const _Float16* __restrict__ Wv,
            const float* __restrict__ bq, const float* __restrict__ bk,
            const float* __restrict__ bv, float* __restrict__ Fatt_g) {
  __shared__ __align__(16) _Float16 smem[69120];     // 138,240 B
  _Float16* Qt = smem;
  _Float16* Kt = smem + 16896;
  _Float16* Rt = smem + 33792;                       // reused as Vm after phase 1
  _Float16* Vm = smem + 33792;
  _Float16* Dt = smem + 52224;                       // reused as Af/As after phase 2
  float*    Af = reinterpret_cast<float*>(smem + 52224);
  _Float16* As = smem + 52224 + 8704;

  const int tid  = threadIdx.x;
  const int lane = tid & 31;
  const int wid  = tid >> 5;
  const int b    = blockIdx.x;
  const int col0 = lane & 15;
  const int rhi  = (lane >> 4) << 3;
  const float8_t fzero = {0.f,0.f,0.f,0.f,0.f,0.f,0.f,0.f};

  // Stage pooled features into padded LDS
  const _Float16* gR = Rt_g + (size_t)b * (NT * NC);
  const _Float16* gD = Dt_g + (size_t)b * (NT * NC);
  for (int i = tid; i < 2048; i += 256) {
    int row = i >> 5, c8 = (i & 31) << 3;
    *reinterpret_cast<half8_t*>(Rt + row * LDP + c8) =
        *reinterpret_cast<const half8_t*>(gR + row * 256 + c8);
    *reinterpret_cast<half8_t*>(Dt + row * LDP + c8) =
        *reinterpret_cast<const half8_t*>(gD + row * 256 + c8);
  }
  __syncthreads();

  // Phase 1: Qt[t][o] = sum_c Rt[t][c]*Wq[o][c] + bq[o]; Kt likewise from Dt/Wk.
  // A-frag = Rt/Dt rows (LDS), B-frag = W row-major [o][c] (global, L2-hot).
  {
    const int mt = wid >> 1;              // token tile 0..3
    const int nb = (wid & 1) * 8;         // 8 out-channel tiles
    float8_t qa[8], ka[8];
#pragma unroll
    for (int j = 0; j < 8; ++j) { qa[j] = fzero; ka[j] = fzero; }
    for (int k = 0; k < 256; k += 32) {
      half16_t ar = frag_a(Rt + mt * 16 * LDP + k, LDP);
      half16_t ad = frag_a(Dt + mt * 16 * LDP + k, LDP);
#pragma unroll
      for (int j = 0; j < 8; ++j) {
        half16_t wq = frag_b(Wq + (nb + j) * 16 * 256 + k, 256);
        qa[j] = wmma_f16(ar, wq, qa[j]);
        half16_t wk = frag_b(Wk + (nb + j) * 16 * 256 + k, 256);
        ka[j] = wmma_f16(ad, wk, ka[j]);
      }
    }
#pragma unroll
    for (int j = 0; j < 8; ++j) {
      int col = (nb + j) * 16 + col0;
      float bqv = bq[col], bkv = bk[col];
#pragma unroll
      for (int r = 0; r < 8; ++r) {
        int row = mt * 16 + rhi + r;
        Qt[row * LDP + col] = (_Float16)(qa[j][r] + bqv);
        Kt[row * LDP + col] = (_Float16)(ka[j][r] + bkv);
      }
    }
  }
  __syncthreads();

  // Phase 2: Vm[o][t] = sum_c Wv[o][c]*Dt[t][c] + bv[o]   (writes over Rt region)
  for (int mi = 0; mi < 2; ++mi) {
    const int mt = wid * 2 + mi;          // out-channel tile 0..15
    float8_t va[4];
#pragma unroll
    for (int n = 0; n < 4; ++n) va[n] = fzero;
    for (int k = 0; k < 256; k += 32) {
      half16_t aw = frag_a(Wv + mt * 16 * 256 + k, 256);
#pragma unroll
      for (int n = 0; n < 4; ++n) {
        half16_t bd = frag_b(Dt + n * 16 * LDP + k, LDP);
        va[n] = wmma_f16(aw, bd, va[n]);
      }
    }
#pragma unroll
    for (int n = 0; n < 4; ++n)
#pragma unroll
      for (int r = 0; r < 8; ++r) {
        int row = mt * 16 + rhi + r;
        Vm[row * LDV + n * 16 + col0] = (_Float16)(va[n][r] + bv[row]);
      }
  }
  __syncthreads();

  // Phase 3: Af[t][s] = sum_c Qt[t][c]*Kt[s][c]   (writes over Dt region)
  for (int ti = 0; ti < 2; ++ti) {
    const int tile = wid * 2 + ti;        // 0..15
    const int mt = tile >> 2, nt = tile & 3;
    float8_t aa = fzero;
    for (int k = 0; k < 256; k += 32) {
      half16_t aq = frag_a(Qt + mt * 16 * LDP + k, LDP);
      half16_t bk2 = frag_b(Kt + nt * 16 * LDP + k, LDP);
      aa = wmma_f16(aq, bk2, aa);
    }
#pragma unroll
    for (int r = 0; r < 8; ++r)
      Af[(mt * 16 + rhi + r) * LDAF + nt * 16 + col0] = aa[r];
  }
  __syncthreads();

  // Phase 4: row softmax in f32, emit f16 As[t][s]
  if (tid < 64) {
    float mx = -3.4e38f;
    for (int s = 0; s < 64; ++s) mx = fmaxf(mx, Af[tid * LDAF + s]);
    float sum = 0.f;
    for (int s = 0; s < 64; ++s) {
      float e = __expf(Af[tid * LDAF + s] - mx);
      Af[tid * LDAF + s] = e;
      sum += e;
    }
    float rs = 1.f / sum;
    for (int s = 0; s < 64; ++s)
      As[tid * LDV + s] = (_Float16)(Af[tid * LDAF + s] * rs);
  }
  __syncthreads();

  // Phase 5: Fatt[c][t] = sum_s Vm[c][s]*As[t][s]  -> global f32 [b][c][t]
  float* outb = Fatt_g + (size_t)b * (NC * NT);
  for (int mi = 0; mi < 2; ++mi) {
    const int mt = wid * 2 + mi;
    float8_t fa[4];
#pragma unroll
    for (int n = 0; n < 4; ++n) fa[n] = fzero;
    for (int k = 0; k < 64; k += 32) {
      half16_t av = frag_a(Vm + mt * 16 * LDV + k, LDV);
#pragma unroll
      for (int n = 0; n < 4; ++n) {
        half16_t ba = frag_b(As + n * 16 * LDV + k, LDV);
        fa[n] = wmma_f16(av, ba, fa[n]);
      }
    }
#pragma unroll
    for (int n = 0; n < 4; ++n)
#pragma unroll
      for (int r = 0; r < 8; ++r)
        outb[(mt * 16 + rhi + r) * NT + n * 16 + col0] = fa[n][r];
  }
}

// ---------------- K3: bilinear 8->64 upsample (half-pixel) + blend ------------
__global__ void __launch_bounds__(256)
upsample_blend_kernel(const float* __restrict__ Fatt_g, const float* __restrict__ F_rgb,
                      const float* __restrict__ alpha_p, float* __restrict__ out) {
  __shared__ float Ft[64];
  const int bc = blockIdx.x;
  const int tid = threadIdx.x;
  if (tid < 64) Ft[tid] = Fatt_g[(size_t)bc * 64 + tid];
  __syncthreads();
  const float al = alpha_p[0];
  const float om = 1.f - al;
  const float* rgb = F_rgb + (size_t)bc * (NH * NW);
  float* o = out + (size_t)bc * (NH * NW);
#pragma unroll
  for (int i = 0; i < 16; ++i) {
    int p = tid + (i << 8);
    int y = p >> 6, x = p & 63;
    float sy = y * 0.125f - 0.4375f;
    float fy = floorf(sy);
    float wy = sy - fy;
    int y0 = (int)fy, y1 = y0 + 1;
    y0 = y0 < 0 ? 0 : (y0 > 7 ? 7 : y0);
    y1 = y1 < 0 ? 0 : (y1 > 7 ? 7 : y1);
    float sx = x * 0.125f - 0.4375f;
    float fx = floorf(sx);
    float wx = sx - fx;
    int x0 = (int)fx, x1 = x0 + 1;
    x0 = x0 < 0 ? 0 : (x0 > 7 ? 7 : x0);
    x1 = x1 < 0 ? 0 : (x1 > 7 ? 7 : x1);
    float v0 = Ft[y0 * 8 + x0] * (1.f - wx) + Ft[y0 * 8 + x1] * wx;
    float v1 = Ft[y1 * 8 + x0] * (1.f - wx) + Ft[y1 * 8 + x1] * wx;
    float v  = v0 * (1.f - wy) + v1 * wy;
    o[p] = al * v + om * rgb[p];
  }
}

// ---------------- host-side launcher ----------------
extern "C" void kernel_launch(void* const* d_in, const int* in_sizes, int n_in,
                              void* d_out, int out_size, void* d_ws, size_t ws_size,
                              hipStream_t stream) {
  (void)in_sizes; (void)n_in; (void)out_size; (void)ws_size;
  const float* F_rgb = (const float*)d_in[0];
  const float* F_d   = (const float*)d_in[1];
  const float* Wq    = (const float*)d_in[2];
  const float* bq    = (const float*)d_in[3];
  const float* Wk    = (const float*)d_in[4];
  const float* bk    = (const float*)d_in[5];
  const float* Wv    = (const float*)d_in[6];
  const float* bv    = (const float*)d_in[7];
  const float* alpha = (const float*)d_in[8];
  float* out = (float*)d_out;

  // workspace layout
  _Float16* Wq_h = (_Float16*)d_ws;                  // 65536 halves each
  _Float16* Wk_h = Wq_h + 65536;
  _Float16* Wv_h = Wk_h + 65536;
  _Float16* Rt_g = Wv_h + 65536;                     // [32][64][256] f16
  _Float16* Dt_g = Rt_g + (size_t)NB * NT * NC;      // [32][64][256] f16
  float*    Fatt = (float*)(Dt_g + (size_t)NB * NT * NC);  // [32][256][64] f32

  cvt_f32_to_f16<<<256, 256, 0, stream>>>(Wq, Wq_h, NC * NC);
  cvt_f32_to_f16<<<256, 256, 0, stream>>>(Wk, Wk_h, NC * NC);
  cvt_f32_to_f16<<<256, 256, 0, stream>>>(Wv, Wv_h, NC * NC);

  pool_kernel<<<dim3(NB * NC, 2), 64, 0, stream>>>(F_rgb, F_d, Rt_g, Dt_g);

  attn_kernel<<<NB, 256, 0, stream>>>(Rt_g, Dt_g, Wq_h, Wk_h, Wv_h, bq, bk, bv, Fatt);

  upsample_blend_kernel<<<NB * NC, 256, 0, stream>>>(Fatt, F_rgb, alpha, out);
}
